// SelfAttention_60696477827263
// MI455X (gfx1250) — matrix operand order repaired and checked by reading, not compiled, not run on hardware
//
#include <hip/hip_runtime.h>
#include <stdint.h>

#define DEVFN __device__ __forceinline__

typedef __attribute__((ext_vector_type(8)))  float          v8f;
typedef __attribute__((ext_vector_type(16))) __bf16         v16bf;
typedef __attribute__((ext_vector_type(8)))  __bf16         v8bf;
typedef __attribute__((ext_vector_type(8)))  unsigned short u16x8;
typedef __attribute__((ext_vector_type(4)))  unsigned short u16x4;

constexpr int NB   = 4;
constexpr int SEQL = 2048;
constexpr int EMB  = 1024;

// ---------------- helpers ----------------

DEVFN uint16_t f2bf(float x) {  // round-to-nearest-even fp32 -> bf16 bits
  union { float f; uint32_t u; } v; v.f = x;
  uint32_t r = v.u + 0x7FFFu + ((v.u >> 16) & 1u);
  return (uint16_t)(r >> 16);
}

// A fragment, 16x32 bf16 (MxK), A row-major with leading dim lda.
// ISA layout: lanes 0-15 = rows, K 0-7 & 16-23; lanes 16-31 = rows, K 8-15 & 24-31.
DEVFN v16bf load_a_frag(const uint16_t* __restrict__ Arow0, int lda, int k, int lane) {
  const int half = lane >> 4, r = lane & 15;
  const uint16_t* p = Arow0 + (size_t)r * lda + k + 8 * half;
  v8bf lo = *(const v8bf*)(p);        // K = k+8h .. +7   -> VGPR 0..3
  v8bf hi = *(const v8bf*)(p + 16);   // K = k+16+8h ..   -> VGPR 4..7
  return __builtin_shufflevector(lo, hi, 0,1,2,3,4,5,6,7,8,9,10,11,12,13,14,15);
}

// B fragment, 32x16 bf16 (KxN), sourced from Bt (row n of Bt = column n of B),
// lanes 0-15 hold K=0..7 & 16..23 of column n, lanes 16-31 hold K=8..15 & 24..31.
DEVFN v16bf load_b_frag(const uint16_t* __restrict__ Btrow0, int ldb, int k, int lane) {
  const int half = lane >> 4, r = lane & 15;
  const uint16_t* p = Btrow0 + (size_t)r * ldb + k + 8 * half;
  v8bf lo = *(const v8bf*)(p);
  v8bf hi = *(const v8bf*)(p + 16);
  return __builtin_shufflevector(lo, hi, 0,1,2,3,4,5,6,7,8,9,10,11,12,13,14,15);
}

DEVFN v8f wmma_bf16(v16bf a, v16bf b, v8f c) {
  return __builtin_amdgcn_wmma_f32_16x16x32_bf16(false, a, false, b, (short)0, c,
                                                 false, false);
}

// ---------------- fp32 -> bf16 conversion ----------------

__global__ __launch_bounds__(256) void cvt_f32_bf16(const float* __restrict__ in,
                                                    uint16_t* __restrict__ out, int n4) {
  int i = blockIdx.x * 256 + threadIdx.x;
  if (i >= n4) return;
  float4 v = ((const float4*)in)[i];
  u16x4 o;
  o[0] = f2bf(v.x); o[1] = f2bf(v.y); o[2] = f2bf(v.z); o[3] = f2bf(v.w);
  ((u16x4*)out)[i] = o;
}

// ---------------- generic NT GEMM: C = A * Bt^T ----------------
// A:[M,Kd] bf16 row-major, Bt:[N,Kd] bf16 row-major. grid.z batches via strides.
// Block = 256 thr = 8 waves -> 128(M) x 64(N) macro tile; wave tile 16x64.
// EPI: 0 = bf16 row-major, 1 = bf16 transposed store (Ct[n*ldc+m]), 2 = f32 + bias.

template <int EPI>
__global__ __launch_bounds__(256) void gemm_bf16_nt(
    const uint16_t* __restrict__ A, size_t sA, int lda,
    const uint16_t* __restrict__ Bt, size_t sB, int ldb,
    void* __restrict__ Cv, size_t sC, int ldc,
    int Kd, const float* __restrict__ bias) {
  const int tid = threadIdx.x, lane = tid & 31, wave = tid >> 5;
  const int z = blockIdx.z;
  const int m0 = blockIdx.y * 128 + wave * 16;
  const int n0 = blockIdx.x * 64;
  const uint16_t* Az = A + (size_t)z * sA + (size_t)m0 * lda;
  const uint16_t* Bz = Bt + (size_t)z * sB;

  v8f acc[4];
#pragma unroll
  for (int f = 0; f < 4; ++f) acc[f] = 0.0f;

  for (int k = 0; k < Kd; k += 32) {
    v16bf a = load_a_frag(Az, lda, k, lane);
#pragma unroll
    for (int f = 0; f < 4; ++f) {
      v16bf b = load_b_frag(Bz + (size_t)(n0 + 16 * f) * ldb, ldb, k, lane);
      acc[f] = wmma_bf16(a, b, acc[f]);
    }
  }

  // C fragment layout: VGPR i, lanes 0-15 -> (m0+i, n0f+lane); lanes 16-31 -> (m0+8+i,...)
  const int c = lane & 15, mh = (lane >> 4) * 8;
  if constexpr (EPI == 0) {
    uint16_t* C = (uint16_t*)Cv + (size_t)z * sC;
#pragma unroll
    for (int f = 0; f < 4; ++f)
#pragma unroll
      for (int i = 0; i < 8; ++i)
        C[(size_t)(m0 + mh + i) * ldc + (n0 + 16 * f + c)] = f2bf(acc[f][i]);
  } else if constexpr (EPI == 1) {
    uint16_t* C = (uint16_t*)Cv + (size_t)z * sC;
#pragma unroll
    for (int f = 0; f < 4; ++f) {
      u16x8 v;
#pragma unroll
      for (int i = 0; i < 8; ++i) v[i] = f2bf(acc[f][i]);
      *(u16x8*)(C + (size_t)(n0 + 16 * f + c) * ldc + m0 + mh) = v;  // contiguous in m
    }
  } else {
    float* C = (float*)Cv + (size_t)z * sC;
#pragma unroll
    for (int f = 0; f < 4; ++f) {
      int col = n0 + 16 * f + c;
      float b = bias[col];
#pragma unroll
      for (int i = 0; i < 8; ++i)
        C[(size_t)(m0 + mh + i) * ldc + col] = acc[f][i] + b;
    }
  }
}

// ---------------- fused scores + mask + softmax ----------------
// grid = (SEQ/32, NB). Block 256 = 8 waves. Each block: 32 query rows vs all 2048 keys.
// S strip 32x2048 fp32 lives in LDS (256 KB < 320 KB/WGP). Emits normalized bf16 P.

__global__ __launch_bounds__(256) void attn_scores_softmax(
    const uint16_t* __restrict__ Q, const uint16_t* __restrict__ K,
    const int* __restrict__ mask, uint16_t* __restrict__ P) {
  extern __shared__ float smem[];
  float* S      = smem;                 // 32 * 2048
  float* red    = smem + 32 * SEQL;     // 256
  float* rowmax = red + 256;            // 32
  float* rowsum = rowmax + 32;          // 32

  const int b   = blockIdx.y;
  const int qr0 = blockIdx.x * 32;
  const uint16_t* Qb = Q + (size_t)b * SEQL * EMB;
  const uint16_t* Kb = K + (size_t)b * SEQL * EMB;
  const int* Mb      = mask + (size_t)b * SEQL * SEQL;
  uint16_t* Pb       = P + (size_t)b * SEQL * SEQL;

  const int tid = threadIdx.x, lane = tid & 31, wave = tid >> 5;
  const int c = lane & 15, mh = (lane >> 4) * 8;

  // ---- S = (Q Kt)/sqrt(d) with mask, written into LDS ----
#pragma unroll
  for (int mt = 0; mt < 2; ++mt) {
    const int m0 = mt * 16;  // local row base
    for (int nt = 0; nt < 16; ++nt) {
      const int n0 = wave * 256 + nt * 16;  // key column base
      v8f acc = 0.0f;
      for (int k = 0; k < EMB; k += 32) {
        v16bf a  = load_a_frag(Qb + (size_t)(qr0 + m0) * EMB, EMB, k, lane);
        v16bf bb = load_b_frag(Kb + (size_t)n0 * EMB, EMB, k, lane);
        acc = wmma_bf16(a, bb, acc);
      }
#pragma unroll
      for (int i = 0; i < 8; ++i) {
        const int r   = m0 + mh + i;     // local row 0..31
        const int col = n0 + c;          // global key col
        float s = acc[i] * 0.03125f;     // 1/sqrt(1024)
        if (Mb[(size_t)(qr0 + r) * SEQL + col] == 0) s = -1e20f;
        S[r * SEQL + col] = s;
      }
    }
  }
  __syncthreads();

  // ---- row softmax over the 32x2048 strip: 8 threads per row ----
  const int r = tid >> 3, sub = tid & 7;
  float mx = -3.4e38f;
  for (int cc = sub; cc < SEQL; cc += 8) mx = fmaxf(mx, S[r * SEQL + cc]);
  red[tid] = mx;
  __syncthreads();
  if (sub == 0) {
    float m2 = red[tid];
#pragma unroll
    for (int j = 1; j < 8; ++j) m2 = fmaxf(m2, red[tid + j]);
    rowmax[r] = m2;
  }
  __syncthreads();
  const float rm = rowmax[r];
  float sm = 0.0f;
  for (int cc = sub; cc < SEQL; cc += 8) sm += __expf(S[r * SEQL + cc] - rm);
  red[tid] = sm;
  __syncthreads();
  if (sub == 0) {
    float s2 = 0.0f;
#pragma unroll
    for (int j = 0; j < 8; ++j) s2 += red[tid + j];
    rowsum[r] = s2;
  }
  __syncthreads();

  // ---- normalize + emit bf16 P, 16B per thread per row ----
  const int c0 = tid * 8;
  for (int rr = 0; rr < 32; ++rr) {
    const float m2 = rowmax[rr], inv = 1.0f / rowsum[rr];
    u16x8 v;
#pragma unroll
    for (int j = 0; j < 8; ++j)
      v[j] = f2bf(__expf(S[rr * SEQL + c0 + j] - m2) * inv);
    *(u16x8*)(Pb + (size_t)(qr0 + rr) * SEQL + c0) = v;
  }
}

// ---------------- launcher ----------------

extern "C" void kernel_launch(void* const* d_in, const int* in_sizes, int n_in,
                              void* d_out, int out_size, void* d_ws, size_t ws_size,
                              hipStream_t stream) {
  const float* values = (const float*)d_in[0];
  const float* keys   = (const float*)d_in[1];
  const float* query  = (const float*)d_in[2];
  const int*   mask   = (const int*)d_in[3];
  const float* Wv     = (const float*)d_in[4];
  const float* Wk     = (const float*)d_in[5];
  const float* Wq     = (const float*)d_in[6];
  const float* Wo     = (const float*)d_in[7];
  const float* bo     = (const float*)d_in[8];
  float* out          = (float*)d_out;

  char* ws = (char*)d_ws;
  size_t off = 0;
  auto alloc = [&](size_t bytes) {
    void* p = ws + off;
    off += (bytes + 255) & ~(size_t)255;
    return p;
  };
  const size_t WB = (size_t)EMB * EMB * 2;        // 2 MB
  const size_t XB = (size_t)NB * SEQL * EMB * 2;  // 16 MB
  uint16_t* wV = (uint16_t*)alloc(WB);
  uint16_t* wK = (uint16_t*)alloc(WB);
  uint16_t* wQ = (uint16_t*)alloc(WB);
  uint16_t* wO = (uint16_t*)alloc(WB);
  uint16_t* vb = (uint16_t*)alloc(XB);
  uint16_t* kb = (uint16_t*)alloc(XB);
  uint16_t* qb = (uint16_t*)alloc(XB);
  uint16_t* Qp = (uint16_t*)alloc(XB);
  uint16_t* Kp = (uint16_t*)alloc(XB);
  uint16_t* Vt = (uint16_t*)alloc(XB);                          // [b][d][s]
  uint16_t* Pm = (uint16_t*)alloc((size_t)NB * SEQL * SEQL * 2);// 32 MB
  uint16_t* Ob = (uint16_t*)alloc(XB);

  // 1) fp32 -> bf16 conversions
  const int nW4 = EMB * EMB / 4;           // 262144
  const int nX4 = NB * SEQL * EMB / 4;     // 2097152
  cvt_f32_bf16<<<nW4 / 256, 256, 0, stream>>>(Wv, wV, nW4);
  cvt_f32_bf16<<<nW4 / 256, 256, 0, stream>>>(Wk, wK, nW4);
  cvt_f32_bf16<<<nW4 / 256, 256, 0, stream>>>(Wq, wQ, nW4);
  cvt_f32_bf16<<<nW4 / 256, 256, 0, stream>>>(Wo, wO, nW4);
  cvt_f32_bf16<<<nX4 / 256, 256, 0, stream>>>(values, vb, nX4);
  cvt_f32_bf16<<<nX4 / 256, 256, 0, stream>>>(keys, kb, nX4);
  cvt_f32_bf16<<<nX4 / 256, 256, 0, stream>>>(query, qb, nX4);

  const size_t sX = (size_t)SEQL * EMB;  // per-batch elements of [SEQ,EMB]
  dim3 blk(256);
  dim3 gProj(EMB / 64, SEQL / 128, NB);  // (16,16,4)

  // 2) projections: Qp = qb Wq^T, Kp = kb Wk^T, Vt = (vb Wv^T)^T
  gemm_bf16_nt<0><<<gProj, blk, 0, stream>>>(qb, sX, EMB, wQ, 0, EMB,
                                             Qp, sX, EMB, EMB, nullptr);
  gemm_bf16_nt<0><<<gProj, blk, 0, stream>>>(kb, sX, EMB, wK, 0, EMB,
                                             Kp, sX, EMB, EMB, nullptr);
  gemm_bf16_nt<1><<<gProj, blk, 0, stream>>>(vb, sX, EMB, wV, 0, EMB,
                                             Vt, sX, SEQL, EMB, nullptr);

  // 3) fused QK^T + mask + softmax -> Pm (bf16, normalized)
  const size_t smemBytes = (size_t)(32 * SEQL + 256 + 64) * sizeof(float);  // ~257 KB
  attn_scores_softmax<<<dim3(SEQL / 32, NB), blk, smemBytes, stream>>>(Qp, Kp, mask, Pm);

  // 4) Ob = Pm @ V  (B operand = Vt rows, K-contiguous)
  dim3 gPV(EMB / 64, SEQL / 128, NB);
  gemm_bf16_nt<0><<<gPV, blk, 0, stream>>>(Pm, (size_t)SEQL * SEQL, SEQL,
                                           Vt, sX, SEQL,
                                           Ob, sX, EMB, SEQL, nullptr);

  // 5) out = Ob @ Wo^T + bo  (fp32 output)
  gemm_bf16_nt<2><<<gProj, blk, 0, stream>>>(Ob, sX, EMB, wO, 0, EMB,
                                             out, sX, EMB, EMB, bo);
}
